// StrideAwareWiredCfCCell_4294967296464
// MI455X (gfx1250) — matrix operand order repaired and checked by reading, not compiled, not run on hardware
//
#include <hip/hip_runtime.h>

// ---------------------------------------------------------------------------
// CfC cell step for MI455X (gfx1250, wave32).
// GEMM chain via v_wmma_f32_16x16x32_f16 (fp32 accum), double-buffered LDS,
// async global->LDS staging (GLOBAL_LOAD_ASYNC_TO_LDS_B128 + s_wait_asynccnt),
// compile-time-specialized epilogues.
// ---------------------------------------------------------------------------

typedef __attribute__((ext_vector_type(16))) _Float16 v16h;
typedef __attribute__((ext_vector_type(8)))  float    v8f;
typedef int v4i_ __attribute__((vector_size(16)));   // matches builtin pointee

#define B_ 1024
#define D_ 512
#define U_ 1024
#define H_ 2048

#define TM 128     // block tile M
#define TN 128     // block tile N
#define KB 64      // K chunk per stage (= 2 WMMA K-steps)
#define PA 72      // LDS row pitch (halfs) for A tile
#define PB 72      // LDS row pitch (halfs) for B^T tile

#if defined(__gfx1250__) && __has_builtin(__builtin_amdgcn_global_load_async_to_lds_b128)
#define HAVE_ASYNC_LDS 1
#else
#define HAVE_ASYNC_LDS 0
#endif

#define ASYNC_GPTR(p) ((__attribute__((address_space(1))) v4i_*)(p))
#define ASYNC_LPTR(p) ((__attribute__((address_space(3))) v4i_*)(p))

union FragU  { uint4 u[2]; v16h v; };
union HPack  { uint4 q[2]; _Float16 h[16]; };

__device__ __forceinline__ void wait_async_lds() {
#if HAVE_ASYNC_LDS
#if __has_builtin(__builtin_amdgcn_s_wait_asynccnt)
  __builtin_amdgcn_s_wait_asynccnt(0);
#else
  asm volatile("s_wait_asynccnt 0x0" ::: "memory");
#endif
#endif
}

// ---- fp32 -> f16 with optional masking -------------------------------------
// mode 0: plain convert; mode 1: per-column mask broadcast (mask[cols]);
// mode 2: full elementwise mask (mask[n]).
__global__ void cvt_mask_kernel(const float* __restrict__ src,
                                const float* __restrict__ mask,
                                _Float16* __restrict__ dst,
                                int n, int cols, int mode) {
  int i = blockIdx.x * blockDim.x + threadIdx.x;
  int stride = gridDim.x * blockDim.x;
  for (; i < n; i += stride) {
    float v = src[i];
    if (mode == 1)      v *= mask[i % cols];
    else if (mode == 2) v *= mask[i];
    dst[i] = (_Float16)v;
  }
}

// ---- generic f16 WMMA GEMM, compile-time fused epilogue --------------------
// C[M,N] = A[M,K] @ W[K,N] (+ accin) (+ bias[n]);
// ACT: 0=none 1=lecun_tanh 2=sigmoid; OUTF -> fp32 store, OUTH -> f16 store.
template <bool ACC, bool BIAS, int ACT, bool OUTF, bool OUTH>
__global__ void __launch_bounds__(256)
gemm_f16_wmma(const _Float16* __restrict__ A,
              const _Float16* __restrict__ W,
              int M, int N, int K,
              const float* __restrict__ accin,
              const float* __restrict__ bias,
              float* __restrict__ outf,
              _Float16* __restrict__ outh) {
  __shared__ _Float16 lsA[2][TM * PA];
  __shared__ _Float16 lsB[2][TN * PB];

  const int tid  = threadIdx.x;
  const int lane = tid & 31;
  const int w    = tid >> 5;   // wave 0..7
  const int wm   = w >> 2;     // 0..1 : wave row (64 rows)
  const int wn   = w & 3;      // 0..3 : wave col (32 cols)
  const int hi   = lane >> 4;  // lane half
  const int lm   = lane & 15;

  const int rowBase = blockIdx.y * TM;
  const int colBase = blockIdx.x * TN;

  v8f acc[4][2] = {};

  // ---- staging helpers ----
  auto stageA = [&](int buf, int k0) {
    int r  = tid >> 1;                 // 0..127
    int h2 = tid & 1;                  // 0..1  (32 halfs = 64B each)
    const _Float16* gsrc = A + (size_t)(rowBase + r) * K + k0 + h2 * 32;
    _Float16*       ldst = &lsA[buf][r * PA + h2 * 32];
#if HAVE_ASYNC_LDS
    __builtin_amdgcn_global_load_async_to_lds_b128(ASYNC_GPTR(gsrc), ASYNC_LPTR(ldst), 0, 0);
    __builtin_amdgcn_global_load_async_to_lds_b128(ASYNC_GPTR(gsrc), ASYNC_LPTR(ldst), 16, 0);
    __builtin_amdgcn_global_load_async_to_lds_b128(ASYNC_GPTR(gsrc), ASYNC_LPTR(ldst), 32, 0);
    __builtin_amdgcn_global_load_async_to_lds_b128(ASYNC_GPTR(gsrc), ASYNC_LPTR(ldst), 48, 0);
#else
    const uint4* s4 = (const uint4*)gsrc;
    uint4* d4 = (uint4*)ldst;
    uint4 d0 = s4[0], d1 = s4[1], d2 = s4[2], d3 = s4[3];
    d4[0] = d0; d4[1] = d1; d4[2] = d2; d4[3] = d3;
#endif
  };

  auto stageB = [&](int buf, int k0) {
    // W tile [k0..k0+64) x [colBase..colBase+128) -> lsB[n][k] (transposed)
    int kbase = tid >> 3;              // 0..31 (and +32)
    int n0    = (tid & 7) * 16;        // 16 halfs along N
    #pragma unroll
    for (int half = 0; half < 2; half++) {
      int kk = kbase + half * 32;
      HPack p;
      const uint4* src = (const uint4*)(W + (size_t)(k0 + kk) * N + colBase + n0);
      p.q[0] = src[0];
      p.q[1] = src[1];
      #pragma unroll
      for (int j = 0; j < 16; j++)
        lsB[buf][(n0 + j) * PB + kk] = p.h[j];
    }
    if (k0 + KB < K)  // warm the tile after next
      __builtin_prefetch(W + (size_t)(k0 + KB + kbase) * N + colBase + n0, 0, 1);
  };

  stageA(0, 0);
  stageB(0, 0);

  int buf = 0;
  for (int k0 = 0; k0 < K; k0 += KB) {
    wait_async_lds();
    __syncthreads();

    if (k0 + KB < K) {
      stageA(buf ^ 1, k0 + KB);
      stageB(buf ^ 1, k0 + KB);
    }

    // ---- 2 WMMA K-steps from current buffer ----
    #pragma unroll
    for (int kh = 0; kh < 2; kh++) {
      FragU afr[4];
      #pragma unroll
      for (int mt = 0; mt < 4; mt++) {
        int row = wm * 64 + mt * 16 + lm;
        int kb  = kh * 32 + hi * 8;   // lanes<16: K 0-7/16-23 ; >=16: 8-15/24-31
        afr[mt].u[0] = *(const uint4*)&lsA[buf][row * PA + kb];
        afr[mt].u[1] = *(const uint4*)&lsA[buf][row * PA + kb + 16];
      }
      FragU bfr[2];
      #pragma unroll
      for (int nt = 0; nt < 2; nt++) {
        int n  = wn * 32 + nt * 16 + lm;
        int kb = kh * 32 + hi * 8;    // lanes<16: K 0-7/16-23 ; >=16: 8-15/24-31
        bfr[nt].u[0] = *(const uint4*)&lsB[buf][n * PB + kb];
        bfr[nt].u[1] = *(const uint4*)&lsB[buf][n * PB + kb + 16];
      }
      #pragma unroll
      for (int mt = 0; mt < 4; mt++)
        #pragma unroll
        for (int nt = 0; nt < 2; nt++)
          acc[mt][nt] = __builtin_amdgcn_wmma_f32_16x16x32_f16(
              false, afr[mt].v, false, bfr[nt].v,
              (short)0, acc[mt][nt], false, false);
    }
    buf ^= 1;
  }

  // ---- fused epilogue; C layout: VGPR r -> M = r + 8*hi, N = lane&15 ----
  #pragma unroll
  for (int mt = 0; mt < 4; mt++) {
    #pragma unroll
    for (int nt = 0; nt < 2; nt++) {
      #pragma unroll
      for (int r = 0; r < 8; r++) {
        int m = rowBase + wm * 64 + mt * 16 + hi * 8 + r;
        int n = colBase + wn * 32 + nt * 16 + lm;
        size_t idx = (size_t)m * N + n;
        float v = acc[mt][nt][r];
        if constexpr (ACC)  v += accin[idx];
        if constexpr (BIAS) v += bias[n];
        if constexpr (ACT == 1)
          v = 1.7159f * (1.0f - 2.0f / (__expf(1.33333333f * v) + 1.0f));
        else if constexpr (ACT == 2)
          v = 1.0f / (1.0f + __expf(-v));
        if constexpr (OUTF) outf[idx] = v;
        if constexpr (OUTH) outh[idx] = (_Float16)v;
      }
    }
  }
}

// ---- final blend: time gate, h_new, output mask; writes (output, h_new) ----
__global__ void final_blend_kernel(const float* __restrict__ h_prev,
                                   const float* __restrict__ gate,
                                   const float* __restrict__ hcand,
                                   const float* __restrict__ time_kernel,
                                   const float* __restrict__ t_ptr,
                                   const float* __restrict__ output_mask,
                                   float* __restrict__ out, int n, int cols) {
  int i = blockIdx.x * blockDim.x + threadIdx.x;
  if (i >= n) return;
  int c = i & (cols - 1);
  float t_eff = t_ptr[0] * 2.0f;  // STRIDE_LENGTH * TIME_SCALE_FACTOR
  float tg = __expf(-fabsf(t_eff) * __expf(time_kernel[c]));
  float g  = gate[i];
  float hn = h_prev[i] * g + hcand[i] * (1.0f - g) * (1.0f - tg);
  out[i]     = hn * output_mask[c];  // output
  out[n + i] = hn;                   // h_new
}

// ---------------------------------------------------------------------------

extern "C" void kernel_launch(void* const* d_in, const int* in_sizes, int n_in,
                              void* d_out, int out_size, void* d_ws, size_t ws_size,
                              hipStream_t stream) {
  const float* inputs = (const float*)d_in[0];
  const float* h_prev = (const float*)d_in[1];
  const float* t      = (const float*)d_in[2];
  const float* kern   = (const float*)d_in[3];
  const float* rker   = (const float*)d_in[4];
  const float* bbk    = (const float*)d_in[5];
  const float* bbb    = (const float*)d_in[6];
  const float* bbo    = (const float*)d_in[7];
  const float* tker   = (const float*)d_in[8];
  const float* bias   = (const float*)d_in[9];
  const float* rbias  = (const float*)d_in[10];
  const float* gk     = (const float*)d_in[11];
  const float* grk    = (const float*)d_in[12];
  const float* gbias  = (const float*)d_in[13];
  const float* imask  = (const float*)d_in[14];
  const float* rmask  = (const float*)d_in[15];
  const float* omask  = (const float*)d_in[16];
  float* out = (float*)d_out;

  char* ws = (char*)d_ws;
  size_t off = 0;
  auto halloc = [&](size_t elems) -> _Float16* {
    _Float16* p = (_Float16*)(ws + off);
    off += ((elems * 2 + 255) / 256) * 256;
    return p;
  };
  auto falloc = [&](size_t elems) -> float* {
    float* p = (float*)(ws + off);
    off += ((elems * 4 + 255) / 256) * 256;
    return p;
  };

  _Float16* a_in_m = halloc((size_t)B_ * D_);
  _Float16* a_h_m  = halloc((size_t)B_ * U_);
  _Float16* a_in   = halloc((size_t)B_ * D_);
  _Float16* a_h    = halloc((size_t)B_ * U_);
  _Float16* w_k    = halloc((size_t)D_ * H_);
  _Float16* w_r    = halloc((size_t)U_ * H_);
  _Float16* w_b    = halloc((size_t)H_ * H_);
  _Float16* w_bo   = halloc((size_t)H_ * U_);
  _Float16* w_gk   = halloc((size_t)D_ * U_);
  _Float16* w_gr   = halloc((size_t)U_ * U_);
  _Float16* x1     = halloc((size_t)B_ * H_);
  _Float16* x2     = halloc((size_t)B_ * H_);
  float*    c1     = falloc((size_t)B_ * H_);
  float*    hcand  = falloc((size_t)B_ * U_);
  float*    gacc   = falloc((size_t)B_ * U_);
  float*    gateb  = falloc((size_t)B_ * U_);
  (void)ws_size; (void)in_sizes; (void)n_in; (void)out_size;

  const int CT = 256;
  auto cblocks = [](int n) { int b = (n + 255) / 256; return b > 2048 ? 2048 : b; };

  // operand conversion / masking
  cvt_mask_kernel<<<cblocks(B_*D_), CT, 0, stream>>>(inputs, imask, a_in_m, B_*D_, D_, 1);
  cvt_mask_kernel<<<cblocks(B_*U_), CT, 0, stream>>>(h_prev, rmask, a_h_m,  B_*U_, U_, 2);
  cvt_mask_kernel<<<cblocks(B_*D_), CT, 0, stream>>>(inputs, nullptr, a_in, B_*D_, D_, 0);
  cvt_mask_kernel<<<cblocks(B_*U_), CT, 0, stream>>>(h_prev, nullptr, a_h,  B_*U_, U_, 0);
  cvt_mask_kernel<<<cblocks(D_*H_), CT, 0, stream>>>(kern, nullptr, w_k,  D_*H_, 0, 0);
  cvt_mask_kernel<<<cblocks(U_*H_), CT, 0, stream>>>(rker, nullptr, w_r,  U_*H_, 0, 0);
  cvt_mask_kernel<<<cblocks(H_*H_), CT, 0, stream>>>(bbk,  nullptr, w_b,  H_*H_, 0, 0);
  cvt_mask_kernel<<<cblocks(H_*U_), CT, 0, stream>>>(bbo,  nullptr, w_bo, H_*U_, 0, 0);
  cvt_mask_kernel<<<cblocks(D_*U_), CT, 0, stream>>>(gk,   nullptr, w_gk, D_*U_, 0, 0);
  cvt_mask_kernel<<<cblocks(U_*U_), CT, 0, stream>>>(grk,  nullptr, w_gr, U_*U_, 0, 0);

  // GEMM chain
  // G1: c1 = maskedIn @ kernel + bias
  gemm_f16_wmma<false, true, 0, true, false><<<dim3(H_/TN, B_/TM), 256, 0, stream>>>(
      a_in_m, w_k, B_, H_, D_, nullptr, bias, c1, nullptr);
  // G2: x1 = lecun_tanh(c1 + maskedH @ recurrent_kernel)
  gemm_f16_wmma<true, false, 1, false, true><<<dim3(H_/TN, B_/TM), 256, 0, stream>>>(
      a_h_m, w_r, B_, H_, U_, c1, nullptr, nullptr, x1);
  // G3: x2 = lecun_tanh(x1 @ backbone_kernel + backbone_bias)
  gemm_f16_wmma<false, true, 1, false, true><<<dim3(H_/TN, B_/TM), 256, 0, stream>>>(
      x1, w_b, B_, H_, H_, nullptr, bbb, nullptr, x2);
  // G4: hcand = x2 @ backbone_out + recurrent_bias
  gemm_f16_wmma<false, true, 0, true, false><<<dim3(U_/TN, B_/TM), 256, 0, stream>>>(
      x2, w_bo, B_, U_, H_, nullptr, rbias, hcand, nullptr);
  // G5: gacc = inputs @ gate_kernel + gate_bias
  gemm_f16_wmma<false, true, 0, true, false><<<dim3(U_/TN, B_/TM), 256, 0, stream>>>(
      a_in, w_gk, B_, U_, D_, nullptr, gbias, gacc, nullptr);
  // G6: gate = sigmoid(gacc + h_prev @ gate_recurrent_kernel)
  gemm_f16_wmma<true, false, 2, true, false><<<dim3(U_/TN, B_/TM), 256, 0, stream>>>(
      a_h, w_gr, B_, U_, U_, gacc, nullptr, gateb, nullptr);

  // final blend -> (output, h_new)
  final_blend_kernel<<<(B_*U_ + 255) / 256, 256, 0, stream>>>(
      h_prev, gateb, hcand, tker, t, omask, out, B_*U_, U_);
}
